// PILL_89928025243958
// MI455X (gfx1250) — matrix operand order repaired and checked by reading, not compiled, not run on hardware
//
#include <hip/hip_runtime.h>

// ---------------------------------------------------------------------------
// SR-GNN forward on gfx1250 (MI455X): bf16 WMMA for all GEMMs, f32 accumulate.
// Pipeline:
//   1. transpose weights -> bf16 [K][N] layouts
//   2. x = BN1(feat) -> bf16 ; scatter edge src indices into [N][8] slots
//   3. GI = x @ Wih^T + b_ih           (WMMA GEMM, M=65536, N=384, K=128)
//   4. per-graph GRU recurrence (4 steps, 16x128 x 128x384 WMMA in LDS)
//   5. f = BN2(PReLU(x@Wself^T + h@Wneigh^T))   (fused WMMA GEMM)
//   6. feat_u / feat_v / feat_l GEMMs  (WMMA)
//   7. per-graph readout: e -> softmax(16) -> rst_g, pos_g
// ---------------------------------------------------------------------------

typedef __attribute__((ext_vector_type(16))) __bf16 bf16x16;
typedef __attribute__((ext_vector_type(8)))  float  floatx8;

#define N_NODES 65536
#define NGRAPH  4096
#define DIM     128
#define PG      16
#define MAXD    8

// ---- WMMA fragment loaders ------------------------------------------------
// A (16xK row-major, 32-wide K chunk). Per ISA 16-bit A layout:
// lane<16: M=lane, K = {0..7} then {16..23}; lane>=16: M=lane-16, K+8.
static __device__ inline bf16x16 frag_a(const __bf16* base, int ld, int lane) {
  const __bf16* p = base + (lane & 15) * ld + ((lane >> 4) << 3);
  bf16x16 a;
#pragma unroll
  for (int i = 0; i < 8; ++i) { a[i] = p[i]; a[i + 8] = p[i + 16]; }
  return a;
}
// B (32x16 chunk of K-major [K][N] matrix): N = lane%16, K = {0..7},{16..23} (+8 for lane>=16).
static __device__ inline bf16x16 frag_b(const __bf16* base, int ld, int lane) {
  const __bf16* p = base + (lane & 15) + ((lane >> 4) << 3) * ld;
  bf16x16 b;
#pragma unroll
  for (int i = 0; i < 8; ++i) { b[i] = p[i * ld]; b[i + 8] = p[(i + 16) * ld]; }
  return b;
}

// ---- prep kernels ---------------------------------------------------------
__global__ void k_transpose_bf16(const float* __restrict__ W, __bf16* __restrict__ WT,
                                 int outdim, int indim) {
  int t = blockIdx.x * blockDim.x + threadIdx.x;
  if (t >= outdim * indim) return;
  int k = t / outdim, n = t - k * outdim;     // dest [k][n]
  WT[t] = (__bf16)W[n * indim + k];
}

__global__ void k_bn1_x(const float* __restrict__ feat, const float* __restrict__ g,
                        const float* __restrict__ b, const float* __restrict__ m,
                        const float* __restrict__ v, __bf16* __restrict__ xb) {
  int t = blockIdx.x * blockDim.x + threadIdx.x;
  if (t >= N_NODES * DIM) return;
  int d = t & (DIM - 1);
  float s = g[d] * rsqrtf(v[d] + 1e-5f);
  xb[t] = (__bf16)((feat[t] - m[d]) * s + b[d]);
}

__global__ void k_cast_bf16(const float* __restrict__ src, __bf16* __restrict__ dst, int total) {
  int t = blockIdx.x * blockDim.x + threadIdx.x;
  if (t < total) dst[t] = (__bf16)src[t];
}

__global__ void k_init_src(int* __restrict__ srcIdx) {
  int t = blockIdx.x * blockDim.x + threadIdx.x;
  if (t < N_NODES * MAXD) srcIdx[t] = -1;
}

__global__ void k_scatter(const int* __restrict__ esrc, const int* __restrict__ edst,
                          const int* __restrict__ eslot, int* __restrict__ srcIdx, int E) {
  int t = blockIdx.x * blockDim.x + threadIdx.x;
  if (t < E) srcIdx[edst[t] * MAXD + eslot[t]] = esrc[t];
}

// ---- generic one-wave-per-16x16-tile WMMA GEMM: C = A @ Bt (+bias) --------
__global__ void k_gemm_bias(const __bf16* __restrict__ A, int lda,
                            const __bf16* __restrict__ Bt, int ldb,
                            const float* __restrict__ bias,
                            float* __restrict__ C, int ldc,
                            int Mtiles, int Ntiles) {
  int wave = blockIdx.x * (blockDim.x >> 5) + (threadIdx.x >> 5);
  int lane = threadIdx.x & 31;
  if (wave >= Mtiles * Ntiles) return;          // whole-wave uniform exit
  int tm = wave / Ntiles, tn = wave - tm * Ntiles;
  const __bf16* Ab = A + (size_t)tm * 16 * lda;
  const __bf16* Bb = Bt + tn * 16;
  floatx8 acc = {};
  int kch = lda >> 5;
  for (int kc = 0; kc < kch; ++kc) {
    bf16x16 a = frag_a(Ab + kc * 32, lda, lane);
    bf16x16 b = frag_b(Bb + (size_t)(kc * 32) * ldb, ldb, lane);
    acc = __builtin_amdgcn_wmma_f32_16x16x32_bf16(false, a, false, b, (short)0, acc, false, false);
  }
  int col = tn * 16 + (lane & 15);
  float bv = bias ? bias[col] : 0.f;
  int rbase = tm * 16 + ((lane >> 4) << 3);
#pragma unroll
  for (int r = 0; r < 8; ++r)
    C[(size_t)(rbase + r) * ldc + col] = acc[r] + bv;
}

// ---- per-graph GRU recurrence (4 steps) -----------------------------------
__global__ void k_gru(const float* __restrict__ GI, const int* __restrict__ srcIdx,
                      const __bf16* __restrict__ WhhTg, const float* __restrict__ bhh,
                      __bf16* __restrict__ neigh) {
  extern __shared__ __align__(16) char smem[];
  __bf16* sW  = (__bf16*)smem;                                   // 128*384 bf16
  float*  sGH = (float*)(smem + 98304);                          // 16*384 f32
  float*  sH  = (float*)(smem + 98304 + 24576);                  // 16*128 f32
  __bf16* sHb = (__bf16*)(smem + 98304 + 24576 + 8192);          // 16*128 bf16
  int g = blockIdx.x, tid = threadIdx.x, lane = tid & 31, w = tid >> 5;

  // stage W_hh^T into LDS (uint4 copies)
  const uint4* wsrc = (const uint4*)WhhTg;
  uint4* wdst = (uint4*)sW;
  for (int i = tid; i < 98304 / 16; i += 256) wdst[i] = wsrc[i];
  for (int i = tid; i < PG * DIM; i += 256) { sH[i] = 0.f; sHb[i] = (__bf16)0.f; }
  __syncthreads();

  for (int t = 0; t < 4; ++t) {
    // GH = h @ Whh^T + b_hh  : 8 waves x 3 N-tiles x 4 K-chunks
    for (int j = 0; j < 3; ++j) {
      int nt = w + 8 * j;
      floatx8 acc = {};
#pragma unroll
      for (int kc = 0; kc < 4; ++kc) {
        bf16x16 a = frag_a(sHb + kc * 32, DIM, lane);
        bf16x16 b = frag_b(sW + (kc * 32) * 384 + nt * 16, 384, lane);
        acc = __builtin_amdgcn_wmma_f32_16x16x32_bf16(false, a, false, b, (short)0, acc, false, false);
      }
      int col = nt * 16 + (lane & 15);
      float bv = bhh[col];
      int rb = (lane >> 4) << 3;
#pragma unroll
      for (int r = 0; r < 8; ++r) sGH[(rb + r) * 384 + col] = acc[r] + bv;
    }
    __syncthreads();
    // gate math; update h only where a message exists at slot t
    for (int i = tid; i < PG * DIM; i += 256) {
      int n = i >> 7, d = i & (DIM - 1);
      int src = srcIdx[(g * PG + n) * MAXD + t];
      if (src >= 0) {
        const float* gi = GI + (size_t)src * 384;
        float ir = gi[d], iz = gi[128 + d], in = gi[256 + d];
        float hr = sGH[n * 384 + d], hz = sGH[n * 384 + 128 + d], hn = sGH[n * 384 + 256 + d];
        float r = 1.f / (1.f + expf(-(ir + hr)));
        float z = 1.f / (1.f + expf(-(iz + hz)));
        float nn = tanhf(in + r * hn);
        float h = (1.f - z) * nn + z * sH[i];
        sH[i] = h; sHb[i] = (__bf16)h;
      }
    }
    __syncthreads();
  }
  for (int i = tid; i < PG * DIM; i += 256)
    neigh[(size_t)g * (PG * DIM) + i] = sHb[i];
}

// ---- rst = PReLU(x@Ws^T + h@Wn^T); f = BN2(rst) ---------------------------
__global__ void k_rst_f(const __bf16* __restrict__ X, const __bf16* __restrict__ H,
                        const __bf16* __restrict__ Ws, const __bf16* __restrict__ Wn,
                        const float* __restrict__ pa, const float* __restrict__ g2,
                        const float* __restrict__ b2, const float* __restrict__ m2,
                        const float* __restrict__ v2,
                        float* __restrict__ F, __bf16* __restrict__ Fb) {
  int wave = blockIdx.x * 8 + (threadIdx.x >> 5);
  int lane = threadIdx.x & 31;
  int tm = wave >> 3, tn = wave & 7;
  const __bf16* Ax = X + (size_t)tm * 16 * DIM;
  const __bf16* Ah = H + (size_t)tm * 16 * DIM;
  floatx8 acc = {};
#pragma unroll
  for (int kc = 0; kc < 4; ++kc) {
    bf16x16 a = frag_a(Ax + kc * 32, DIM, lane);
    bf16x16 b = frag_b(Ws + (kc * 32) * DIM + tn * 16, DIM, lane);
    acc = __builtin_amdgcn_wmma_f32_16x16x32_bf16(false, a, false, b, (short)0, acc, false, false);
  }
#pragma unroll
  for (int kc = 0; kc < 4; ++kc) {
    bf16x16 a = frag_a(Ah + kc * 32, DIM, lane);
    bf16x16 b = frag_b(Wn + (kc * 32) * DIM + tn * 16, DIM, lane);
    acc = __builtin_amdgcn_wmma_f32_16x16x32_bf16(false, a, false, b, (short)0, acc, false, false);
  }
  int col = tn * 16 + (lane & 15);
  float ap = pa[col], sc = g2[col] * rsqrtf(v2[col] + 1e-5f), mu = m2[col], be = b2[col];
  int rbase = tm * 16 + ((lane >> 4) << 3);
#pragma unroll
  for (int r = 0; r < 8; ++r) {
    float v = acc[r];
    v = v > 0.f ? v : ap * v;
    float f = (v - mu) * sc + be;
    size_t idx = (size_t)(rbase + r) * DIM + col;
    F[idx] = f; Fb[idx] = (__bf16)f;
  }
}

// ---- feat_l = f[last_nodes] @ Wi^T + b_i (gathered rows) ------------------
__global__ void k_feat_l(const __bf16* __restrict__ Fb, const int* __restrict__ last,
                         const __bf16* __restrict__ Wi, const float* __restrict__ bi,
                         float* __restrict__ out) {
  int wave = blockIdx.x * 8 + (threadIdx.x >> 5);
  int lane = threadIdx.x & 31;
  int tm = wave >> 3, tn = wave & 7;
  int grow = last[tm * 16 + (lane & 15)];
  const __bf16* ap0 = Fb + (size_t)grow * DIM + ((lane >> 4) << 3);
  floatx8 acc = {};
#pragma unroll
  for (int kc = 0; kc < 4; ++kc) {
    bf16x16 a;
#pragma unroll
    for (int i = 0; i < 8; ++i) { a[i] = ap0[kc * 32 + i]; a[i + 8] = ap0[kc * 32 + 16 + i]; }
    bf16x16 b = frag_b(Wi + (kc * 32) * DIM + tn * 16, DIM, lane);
    acc = __builtin_amdgcn_wmma_f32_16x16x32_bf16(false, a, false, b, (short)0, acc, false, false);
  }
  int col = tn * 16 + (lane & 15);
  float bv = bi[col];
  int rbase = tm * 16 + ((lane >> 4) << 3);
#pragma unroll
  for (int r = 0; r < 8; ++r)
    out[(size_t)(rbase + r) * DIM + col] = acc[r] + bv;
}

// ---- per-graph attention readout ------------------------------------------
__global__ void k_readout(const float* __restrict__ U, const float* __restrict__ V,
                          const float* __restrict__ L, const float* __restrict__ we,
                          const float* __restrict__ F, const float* __restrict__ pw,
                          float* __restrict__ out) {
  __shared__ float se[PG];
  __shared__ float sa[PG];
  int g = blockIdx.x, tid = threadIdx.x, lane = tid & 31, w = tid >> 5;
  for (int j = 0; j < 4; ++j) {
    int n = w * 4 + j, node = g * PG + n;
    float acc = 0.f;
    for (int d = lane; d < DIM; d += 32) {
      float s = U[(size_t)node * DIM + d] + V[(size_t)g * DIM + d] + L[(size_t)g * DIM + d];
      acc += we[d] / (1.f + expf(-s));
    }
#pragma unroll
    for (int off = 16; off > 0; off >>= 1) acc += __shfl_down(acc, off, 32);
    if (lane == 0) se[n] = acc;
  }
  __syncthreads();
  if (tid == 0) {
    float m = se[0];
    for (int n = 1; n < PG; ++n) m = fmaxf(m, se[n]);
    float s = 0.f;
    for (int n = 0; n < PG; ++n) { float e = expf(se[n] - m); sa[n] = e; s += e; }
    float inv = 1.f / s;
    for (int n = 0; n < PG; ++n) sa[n] *= inv;
  }
  __syncthreads();
  int d = tid;   // blockDim == 128
  float r = 0.f, p = 0.f;
  for (int n = 0; n < PG; ++n) {
    float f = F[((size_t)g * PG + n) * DIM + d];
    r += f * sa[n];
    p += f * pw[g * PG + n];
  }
  out[(size_t)g * DIM + d] = r;
  out[(size_t)NGRAPH * DIM + (size_t)g * DIM + d] = p;
}

// ---------------------------------------------------------------------------
extern "C" void kernel_launch(void* const* d_in, const int* in_sizes, int n_in,
                              void* d_out, int out_size, void* d_ws, size_t ws_size,
                              hipStream_t stream) {
  const float* feat       = (const float*)d_in[0];
  const float* intend     = (const float*)d_in[1];
  const float* pos_w      = (const float*)d_in[2];
  const float* bn1_g      = (const float*)d_in[3];
  const float* bn1_b      = (const float*)d_in[4];
  const float* bn1_m      = (const float*)d_in[5];
  const float* bn1_v      = (const float*)d_in[6];
  const float* gru_w_ih   = (const float*)d_in[7];
  const float* gru_w_hh   = (const float*)d_in[8];
  const float* gru_b_ih   = (const float*)d_in[9];
  const float* gru_b_hh   = (const float*)d_in[10];
  const float* fc_self_w  = (const float*)d_in[11];
  const float* fc_neigh_w = (const float*)d_in[12];
  const float* prelu_a    = (const float*)d_in[13];
  const float* bn2_g      = (const float*)d_in[14];
  const float* bn2_b      = (const float*)d_in[15];
  const float* bn2_m      = (const float*)d_in[16];
  const float* bn2_v      = (const float*)d_in[17];
  const float* fc_u_w     = (const float*)d_in[18];
  const float* fc_v_w     = (const float*)d_in[19];
  const float* fc_v_b     = (const float*)d_in[20];
  const float* fc_i_w     = (const float*)d_in[21];
  const float* fc_i_b     = (const float*)d_in[22];
  const float* fc_e_w     = (const float*)d_in[23];
  const int*   last_nodes = (const int*)d_in[24];
  const int*   edge_src   = (const int*)d_in[25];
  const int*   edge_dst   = (const int*)d_in[26];
  const int*   edge_slot  = (const int*)d_in[27];
  int E = in_sizes[25];

  // workspace carve-up
  char* ws = (char*)d_ws;
  size_t off = 0;
  auto take = [&](size_t bytes) { size_t o = off; off = (off + bytes + 255) & ~(size_t)255; return o; };
  __bf16* WihT   = (__bf16*)(ws + take(128 * 384 * 2));
  __bf16* WhhT   = (__bf16*)(ws + take(128 * 384 * 2));
  __bf16* WselfT = (__bf16*)(ws + take(128 * 128 * 2));
  __bf16* WneighT= (__bf16*)(ws + take(128 * 128 * 2));
  __bf16* WuT    = (__bf16*)(ws + take(128 * 128 * 2));
  __bf16* WvT    = (__bf16*)(ws + take(128 * 128 * 2));
  __bf16* WiT    = (__bf16*)(ws + take(128 * 128 * 2));
  __bf16* x_bf   = (__bf16*)(ws + take((size_t)N_NODES * DIM * 2));
  __bf16* in_bf  = (__bf16*)(ws + take((size_t)NGRAPH * DIM * 2));
  __bf16* h_bf   = (__bf16*)(ws + take((size_t)N_NODES * DIM * 2));
  __bf16* f_bf   = (__bf16*)(ws + take((size_t)N_NODES * DIM * 2));
  float*  GI     = (float*)(ws + take((size_t)N_NODES * 384 * 4));
  float*  f_f32  = (float*)(ws + take((size_t)N_NODES * DIM * 4));
  float*  feat_u = (float*)(ws + take((size_t)N_NODES * DIM * 4));
  float*  feat_v = (float*)(ws + take((size_t)NGRAPH * DIM * 4));
  float*  feat_l = (float*)(ws + take((size_t)NGRAPH * DIM * 4));
  int*    srcIdx = (int*)(ws + take((size_t)N_NODES * MAXD * 4));

  const int T = 256;
  // 1. weight transposes -> bf16 [K][N]
  k_transpose_bf16<<<(128 * 384 + T - 1) / T, T, 0, stream>>>(gru_w_ih, WihT, 384, 128);
  k_transpose_bf16<<<(128 * 384 + T - 1) / T, T, 0, stream>>>(gru_w_hh, WhhT, 384, 128);
  k_transpose_bf16<<<(128 * 128 + T - 1) / T, T, 0, stream>>>(fc_self_w, WselfT, 128, 128);
  k_transpose_bf16<<<(128 * 128 + T - 1) / T, T, 0, stream>>>(fc_neigh_w, WneighT, 128, 128);
  k_transpose_bf16<<<(128 * 128 + T - 1) / T, T, 0, stream>>>(fc_u_w, WuT, 128, 128);
  k_transpose_bf16<<<(128 * 128 + T - 1) / T, T, 0, stream>>>(fc_v_w, WvT, 128, 128);
  k_transpose_bf16<<<(128 * 128 + T - 1) / T, T, 0, stream>>>(fc_i_w, WiT, 128, 128);
  // 2. BN1 + casts + edge scatter
  k_bn1_x<<<(N_NODES * DIM) / T, T, 0, stream>>>(feat, bn1_g, bn1_b, bn1_m, bn1_v, x_bf);
  k_cast_bf16<<<(NGRAPH * DIM) / T, T, 0, stream>>>(intend, in_bf, NGRAPH * DIM);
  k_init_src<<<(N_NODES * MAXD) / T, T, 0, stream>>>(srcIdx);
  k_scatter<<<(E + T - 1) / T, T, 0, stream>>>(edge_src, edge_dst, edge_slot, srcIdx, E);
  // 3. GI = x @ Wih^T + b_ih
  k_gemm_bias<<<(4096 * 24 + 7) / 8, T, 0, stream>>>(x_bf, 128, WihT, 384, gru_b_ih, GI, 384, 4096, 24);
  // 4. GRU recurrence per graph
  k_gru<<<NGRAPH, T, 98304 + 24576 + 8192 + 4096, stream>>>(GI, srcIdx, WhhT, gru_b_hh, h_bf);
  // 5. f = BN2(PReLU(x@Ws^T + h@Wn^T))
  k_rst_f<<<(4096 * 8) / 8, T, 0, stream>>>(x_bf, h_bf, WselfT, WneighT, prelu_a,
                                            bn2_g, bn2_b, bn2_m, bn2_v, f_f32, f_bf);
  // 6. attention inputs
  k_gemm_bias<<<(4096 * 8) / 8, T, 0, stream>>>(f_bf, 128, WuT, 128, nullptr, feat_u, 128, 4096, 8);
  k_gemm_bias<<<(256 * 8) / 8, T, 0, stream>>>(in_bf, 128, WvT, 128, fc_v_b, feat_v, 128, 256, 8);
  k_feat_l<<<(256 * 8) / 8, T, 0, stream>>>(f_bf, last_nodes, WiT, fc_i_b, feat_l);
  // 7. softmax readout -> [rst_g | pos_g]
  k_readout<<<NGRAPH, 128, 0, stream>>>(feat_u, feat_v, feat_l, fc_e_w, f_f32, pos_w, (float*)d_out);
}
